// BiDiGemma_61933428417042
// MI455X (gfx1250) — matrix operand order repaired and checked by reading, not compiled, not run on hardware
//
#include <hip/hip_runtime.h>
#include <math.h>
#include <stdint.h>

// ---------------- model constants ----------------
#define Bc   4
#define Sc   4096
#define Dc   256
#define Hc   4
#define HDc  64
#define Ec   768
#define Mrows (Bc*Sc)          // 16384 tokens
#define NCc  64                // scan chunks
#define CLc  (Sc/NCc)          // 64 steps per chunk

// ---------------- WMMA plumbing -----------------
typedef __attribute__((ext_vector_type(16))) __bf16 bf16x16;
typedef __attribute__((ext_vector_type(8)))  float  f32x8;

union AFrag { bf16x16 v; unsigned u[8]; };

__device__ __forceinline__ unsigned short f2bf(float f){
  union { float f; unsigned u; } x; x.f = f;
  unsigned u = x.u;
  u += 0x7FFFu + ((u >> 16) & 1u);       // round-to-nearest-even
  return (unsigned short)(u >> 16);
}
__device__ __forceinline__ f32x8 wmma_bf16(AFrag a, AFrag b, f32x8 c){
  return __builtin_amdgcn_wmma_f32_16x16x32_bf16(false, a.v, false, b.v,
                                                 (short)0, c, false, false);
}
__device__ __forceinline__ float gelu_f(float x){
  return 0.5f * x * (1.f + erff(x * 0.70710678118654752f));
}
__device__ __forceinline__ float sigmoid_f(float x){ return 1.f / (1.f + expf(-x)); }

// ---------------- CDNA5 async LDS copy (16B per lane) ----------------
#define USE_ASYNC_COPY 1

__device__ __forceinline__ void cp16(const void* g, void* l){
#if USE_ASYNC_COPY
  unsigned lds = (unsigned)(uintptr_t)l;                 // addr[31:0] = LDS offset
  unsigned long long ga = (unsigned long long)(uintptr_t)g;
  asm volatile("global_load_async_to_lds_b128 %0, %1, off"
               :: "v"(lds), "v"(ga) : "memory");
#else
  *(uint4*)l = *(const uint4*)g;
#endif
}
__device__ __forceinline__ void async_fence(){
#if USE_ASYNC_COPY
  asm volatile("s_wait_asynccnt 0" ::: "memory");
#endif
}

// =======================================================================
// GEMM: C[M,N] = act(A[M,K] @ W[K,N] + bias) + res
// A, W pre-converted bf16; C fp32. Block tile 128x64, BK=32, 8 waves,
// wave = 32x32 sub-tile (2x2 WMMA). M%128==0, K%32==0, N%64==0 for all uses.
// Tiles staged to LDS with async b128 copies.
// =======================================================================
#define GBM 128
#define GBN 64
#define GBK 32
#define GPA 40   // A LDS pitch (ushorts): 80B rows -> b128-aligned segments
#define GPB 72   // B LDS pitch (ushorts): 144B rows -> b128-aligned segments

__global__ __launch_bounds__(256) void gemm_bf16_wmma(
    const unsigned short* __restrict__ A, int lda,
    const unsigned short* __restrict__ W, int ldw,
    const float* __restrict__ bias,
    const float* res, int ldres,
    float* __restrict__ C, int ldc,
    int M, int N, int K, int act)
{
  __shared__ unsigned short As[GBM * GPA];   // [row][k]
  __shared__ unsigned short Bs[GBK * GPB];   // [k][n]
  const int tid  = threadIdx.x;
  const int lane = tid & 31;
  const int wave = tid >> 5;
  const int m0 = blockIdx.y * GBM;
  const int n0 = blockIdx.x * GBN;
  const int wm = (wave >> 1) * 32;
  const int wn = (wave & 1) * 32;
  const int half = lane >> 4, l16 = lane & 15;

  // per-thread staging coordinates (8 bf16 = 16B chunks)
  const int rA0 = tid >> 2,           cA0 = (tid & 3) << 3;        // chunk 0
  const int rA1 = (tid + 256) >> 2,   cA1 = cA0;                   // chunk 1
  const int rB  = tid >> 3,           cB  = (tid & 7) << 3;

  f32x8 acc[2][2];
  #pragma unroll
  for (int i = 0; i < 2; ++i)
    #pragma unroll
    for (int j = 0; j < 2; ++j)
      #pragma unroll
      for (int r = 0; r < 8; ++r) acc[i][j][r] = 0.f;

  for (int k0 = 0; k0 < K; k0 += GBK) {
    // ---- async-stage A (128x32) and B (32x64) bf16 tiles ----
    cp16(A + (size_t)(m0 + rA0) * lda + k0 + cA0, &As[rA0 * GPA + cA0]);
    cp16(A + (size_t)(m0 + rA1) * lda + k0 + cA1, &As[rA1 * GPA + cA1]);
    cp16(W + (size_t)(k0 + rB)  * ldw + n0 + cB,  &Bs[rB  * GPB + cB]);
    if (k0 + GBK < K) {
      __builtin_prefetch(A + (size_t)(m0 + rA0) * lda + k0 + GBK + cA0, 0, 1);
      __builtin_prefetch(W + (size_t)(k0 + GBK + rB) * ldw + n0 + cB,  0, 1);
    }
    async_fence();
    __syncthreads();

    AFrag af[2], bfr[2];
    #pragma unroll
    for (int ti = 0; ti < 2; ++ti) {
      const unsigned short* p = &As[(wm + ti * 16 + l16) * GPA];
      #pragma unroll
      for (int v = 0; v < 8; ++v) {
        int kk = (half << 4) + 2 * v;
        af[ti].u[v] = *(const unsigned*)(p + kk);
      }
    }
    #pragma unroll
    for (int tj = 0; tj < 2; ++tj) {
      int n = wn + tj * 16 + l16;
      #pragma unroll
      for (int v = 0; v < 8; ++v) {
        int kk = (half << 4) + 2 * v;
        unsigned lo = Bs[kk * GPB + n];
        unsigned hi = Bs[(kk + 1) * GPB + n];
        bfr[tj].u[v] = lo | (hi << 16);
      }
    }
    #pragma unroll
    for (int ti = 0; ti < 2; ++ti)
      #pragma unroll
      for (int tj = 0; tj < 2; ++tj)
        acc[ti][tj] = wmma_bf16(af[ti], bfr[tj], acc[ti][tj]);
    __syncthreads();
  }

  // epilogue: C layout — vgpr r: row half*8+r, col lane%16
  #pragma unroll
  for (int ti = 0; ti < 2; ++ti)
    #pragma unroll
    for (int tj = 0; tj < 2; ++tj) {
      int col = n0 + wn + tj * 16 + l16;
      if (col >= N) continue;
      #pragma unroll
      for (int r = 0; r < 8; ++r) {
        int row = m0 + wm + ti * 16 + half * 8 + r;
        float v = acc[ti][tj][r];
        if (bias) v += bias[col];
        if (act == 1) v = gelu_f(v);
        if (res) v += res[(size_t)row * ldres + col];
        C[(size_t)row * ldc + col] = v;
      }
    }
}

// =======================================================================
// fp32 -> bf16 conversion (weights once per call, odd activations)
// =======================================================================
__global__ void cvt_kernel(const float* __restrict__ in,
                           unsigned short* __restrict__ o, long n)
{
  long idx = (long)blockIdx.x * 256 + threadIdx.x;
  if (idx < n) o[idx] = f2bf(in[idx]);
}

// =======================================================================
// Prologue: layer_norm(skip) (bf16 out) and z = sigmoid(mean(silu(skip)))
// =======================================================================
__global__ __launch_bounds__(256) void ln_silu_kernel(
    const float* __restrict__ skip, const float* __restrict__ g,
    const float* __restrict__ be, unsigned short* __restrict__ out,
    float* __restrict__ z)
{
  int tok = blockIdx.x, d = threadIdx.x;
  float x = skip[(size_t)tok * Dc + d];
  float s1 = x, s2 = x * x, s3 = x * sigmoid_f(x);
  #pragma unroll
  for (int o = 1; o < 32; o <<= 1) {
    s1 += __shfl_xor(s1, o, 32);
    s2 += __shfl_xor(s2, o, 32);
    s3 += __shfl_xor(s3, o, 32);
  }
  __shared__ float red[3][8];
  if ((threadIdx.x & 31) == 0) {
    int w = threadIdx.x >> 5;
    red[0][w] = s1; red[1][w] = s2; red[2][w] = s3;
  }
  __syncthreads();
  float t1 = 0, t2 = 0, t3 = 0;
  #pragma unroll
  for (int i = 0; i < 8; ++i) { t1 += red[0][i]; t2 += red[1][i]; t3 += red[2][i]; }
  float mean = t1 / (float)Dc;
  float var  = t2 / (float)Dc - mean * mean;
  out[(size_t)tok * Dc + d] = f2bf((x - mean) * rsqrtf(var + 1e-5f) * g[d] + be[d]);
  if (d == 0) z[tok] = sigmoid_f(t3 / (float)Dc);
}

__global__ __launch_bounds__(256) void rms_kernel(
    const float* __restrict__ x, const float* __restrict__ scale,
    unsigned short* __restrict__ out)
{
  int tok = blockIdx.x, d = threadIdx.x;
  float v = x[(size_t)tok * Dc + d];
  float s = v * v;
  #pragma unroll
  for (int o = 1; o < 32; o <<= 1) s += __shfl_xor(s, o, 32);
  __shared__ float red[8];
  if ((threadIdx.x & 31) == 0) red[threadIdx.x >> 5] = s;
  __syncthreads();
  float tot = 0;
  #pragma unroll
  for (int i = 0; i < 8; ++i) tot += red[i];
  float r = rsqrtf(tot / (float)Dc + 1e-6f);
  out[(size_t)tok * Dc + d] = f2bf(v * r * (1.f + scale[d]));
}

// =======================================================================
// depthwise causal conv1d, width 4; dual fp32 + bf16 output
// =======================================================================
__global__ void conv1d_kernel(const float* __restrict__ x,
                              const float* __restrict__ w,
                              const float* __restrict__ b,
                              float* __restrict__ out,
                              unsigned short* __restrict__ outbf, long n)
{
  long idx = (long)blockIdx.x * 256 + threadIdx.x;
  if (idx >= n) return;
  int d = idx & (Dc - 1);
  long bs = idx >> 8;
  int s = (int)(bs % Sc);
  long bb = bs / Sc;
  float acc = b[d];
  #pragma unroll
  for (int i = 0; i < 4; ++i) {
    int ss = s - 3 + i;
    if (ss >= 0) acc += x[((bb * Sc + ss) << 8) + d] * w[i * Dc + d];
  }
  out[idx] = acc;
  outbf[idx] = f2bf(acc);
}

// =======================================================================
// RG-LRU prep (in-place: gate_x buf -> a, x buf -> bterm)
// =======================================================================
__global__ void lru_prep_kernel(float* xb, float* gxb,
                                const float* __restrict__ gab,
                                const float* __restrict__ ap, long n)
{
  long idx = (long)blockIdx.x * 256 + threadIdx.x;
  if (idx >= n) return;
  int d = idx & (Dc - 1);
  long bs = idx >> 8;
  int s = (int)(bs % Sc);
  float x  = xb[idx];
  float gx = sigmoid_f(gxb[idx]);
  float ga = sigmoid_f(gab[idx]);
  float a0 = ap[d];
  float sp = (a0 > 20.f) ? a0 : log1pf(expf(a0));
  float log_a = -8.f * ga * sp;
  float a = expf(log_a);
  float mult = (s == 0) ? 1.f : sqrtf(-expm1f(2.f * log_a));
  gxb[idx] = a;
  xb[idx]  = x * gx * mult;
}

// ----- chunked linear-recurrence scan: h_t = a_t h_{t-1} + b_t -----
__global__ void scan1_kernel(float* a, float* bt, float* cA, float* cB)
{
  int idx = blockIdx.x * 256 + threadIdx.x;     // (b*D+d)*NC + c
  if (idx >= Bc * Dc * NCc) return;
  int c = idx % NCc;
  int bd = idx / NCc;
  int d = bd & (Dc - 1);
  int b = bd >> 8;
  float cum = 1.f, h = 0.f;
  size_t base = ((size_t)b * Sc) * Dc + d;
  for (int i = 0; i < CLc; ++i) {
    size_t off = base + (size_t)(c * CLc + i) * Dc;
    float av = a[off], bv = bt[off];
    cum *= av;
    h = av * h + bv;
    a[off] = cum;
    bt[off] = h;
  }
  cA[idx] = cum; cB[idx] = h;
}
__global__ void scan2_kernel(const float* __restrict__ cA,
                             const float* __restrict__ cB,
                             float* __restrict__ carry)
{
  int idx = blockIdx.x * 256 + threadIdx.x;     // b*D+d
  if (idx >= Bc * Dc) return;
  float cy = 0.f;
  for (int c = 0; c < NCc; ++c) {
    carry[idx * NCc + c] = cy;
    cy = cA[idx * NCc + c] * cy + cB[idx * NCc + c];
  }
}
// h = hloc + cumA*carry;  out_bf16 = h * y
__global__ void scan3_kernel(const float* __restrict__ cumA,
                             const float* __restrict__ hloc,
                             const float* __restrict__ carry,
                             const float* __restrict__ y,
                             unsigned short* __restrict__ obf, long n)
{
  long idx = (long)blockIdx.x * 256 + threadIdx.x;
  if (idx >= n) return;
  int d = idx & (Dc - 1);
  long bs = idx >> 8;
  int s = (int)(bs % Sc);
  int b = (int)(bs / Sc);
  int c = s / CLc;
  float cy = carry[((size_t)(b * Dc + d)) * NCc + c];
  float h = hloc[idx] + cumA[idx] * cy;
  obf[idx] = f2bf(h * y[idx]);
}

// gated product (bf16 out for down-proj GEMM)
__global__ void mul_kernel(const float* __restrict__ a,
                           const float* __restrict__ b,
                           unsigned short* __restrict__ obf, long n)
{
  long idx = (long)blockIdx.x * 256 + threadIdx.x;
  if (idx >= n) return;
  obf[idx] = f2bf(a[idx] * b[idx]);
}

// =======================================================================
// partial RoPE + scale, fp32 in -> bf16 out; one thread per (b,s,h,i<16)
// =======================================================================
__global__ void rope_bf_kernel(const float* __restrict__ x,
                               unsigned short* __restrict__ o,
                               int nh, float scale, long n)
{
  long idx = (long)blockIdx.x * 256 + threadIdx.x;
  if (idx >= n) return;
  int i = idx & 15;
  long r = idx >> 4;
  int h = (int)(r % nh);
  long bs = r / nh;
  int s = (int)(bs % Sc);
  const float* base = x + ((bs * nh) + h) * 64;
  unsigned short* ob = o + ((bs * nh) + h) * 64;
  float inv = powf(10000.f, -(float)i / 16.f);
  float t = (float)s * inv;
  float sn = sinf(t), cs = cosf(t);
  float x1 = base[i], x2 = base[16 + i];
  ob[i]          = f2bf((x1 * cs - x2 * sn) * scale);
  ob[16 + i]     = f2bf((x2 * cs + x1 * sn) * scale);
  ob[32 + 2 * i] = f2bf(base[32 + 2 * i] * scale);
  ob[33 + 2 * i] = f2bf(base[33 + 2 * i] * scale);
}

// =======================================================================
// Sliding-window MQA flash attention (window 256); bf16 q/k/v in, bf16 out.
// One wave per 16-row q tile; QK^T and P*V on WMMA; P via per-wave LDS patch.
// =======================================================================
__global__ __launch_bounds__(256) void attn_kernel(
    const unsigned short* __restrict__ q,   // [B,S,H,64] bf16 (roped, scaled)
    const unsigned short* __restrict__ kk,  // [B,S,64]   bf16 (roped)
    const unsigned short* __restrict__ vv,  // [B,S,64]   bf16
    unsigned short* __restrict__ out)       // [B,S,256]  bf16
{
  __shared__ unsigned short Pld[8 * 16 * 34];
  const int tid = threadIdx.x, lane = tid & 31, wave = tid >> 5;
  const int half = lane >> 4, l16 = lane & 15;
  int t = blockIdx.x * 8 + wave;                // 4096 tiles
  int qt = t & 15;
  int h  = (t >> 4) & 3;
  int nC = (t >> 6) & 15;
  int b  = t >> 10;
  int qlo = qt * 16;

  AFrag qf[2];
  {
    int srow = nC * 256 + qlo + l16;
    const unsigned short* qp = q + (((size_t)b * Sc + srow) * Hc + h) * 64;
    #pragma unroll
    for (int db = 0; db < 2; ++db)
      #pragma unroll
      for (int v = 0; v < 8; ++v) {
        int d = (half << 4) + 2 * v + db * 32;
        qf[db].u[v] = *(const unsigned*)(qp + d);
      }
  }

  float mrow[8], lrow[8], scf[8];
  #pragma unroll
  for (int r = 0; r < 8; ++r) { mrow[r] = -1e30f; lrow[r] = 0.f; }
  f32x8 oacc[4];
  #pragma unroll
  for (int nt = 0; nt < 4; ++nt)
    #pragma unroll
    for (int r = 0; r < 8; ++r) oacc[nt][r] = 0.f;

  unsigned short* P = &Pld[wave * 16 * 34];
  int kbmin = qlo >> 5;
  if (nC == 0 && kbmin < 8) kbmin = 8;
  int kbmax = (qlo + 271) >> 5; if (kbmax > 15) kbmax = 15;

  for (int kb = kbmin; kb <= kbmax; ++kb) {
    f32x8 c2[2];
    #pragma unroll
    for (int kt = 0; kt < 2; ++kt)
      #pragma unroll
      for (int r = 0; r < 8; ++r) c2[kt][r] = 0.f;

    #pragma unroll
    for (int kt = 0; kt < 2; ++kt) {
      int key = kb * 32 + kt * 16 + l16;
      int s = (nC - 1) * 256 + key;
      const unsigned short* kp = (s >= 0) ? kk + ((size_t)b * Sc + s) * 64 : nullptr;
      #pragma unroll
      for (int db = 0; db < 2; ++db) {
        AFrag kf;
        #pragma unroll
        for (int v = 0; v < 8; ++v) {
          int d = (half << 4) + 2 * v + db * 32;
          kf.u[v] = kp ? *(const unsigned*)(kp + d) : 0u;
        }
        c2[kt] = wmma_bf16(qf[db], kf, c2[kt]);
      }
    }
    #pragma unroll
    for (int kt = 0; kt < 2; ++kt) {
      int j = kb * 32 + kt * 16 + l16;
      #pragma unroll
      for (int r = 0; r < 8; ++r) {
        int qi = qlo + half * 8 + r;
        bool valid = (j >= qi) && (j <= qi + 256) && (nC > 0 || j >= 256);
        if (!valid) c2[kt][r] = -1e30f;
      }
    }
    #pragma unroll
    for (int r = 0; r < 8; ++r) {
      float tm = fmaxf(c2[0][r], c2[1][r]);
      #pragma unroll
      for (int o = 1; o < 16; o <<= 1) tm = fmaxf(tm, __shfl_xor(tm, o, 16));
      float mnew = fmaxf(mrow[r], tm);
      float sc = expf(mrow[r] - mnew);
      mrow[r] = mnew; scf[r] = sc;
      float p0 = (c2[0][r] > -5e29f) ? expf(c2[0][r] - mnew) : 0.f;
      float p1 = (c2[1][r] > -5e29f) ? expf(c2[1][r] - mnew) : 0.f;
      c2[0][r] = p0; c2[1][r] = p1;
      float ps = p0 + p1;
      #pragma unroll
      for (int o = 1; o < 16; o <<= 1) ps += __shfl_xor(ps, o, 16);
      lrow[r] = lrow[r] * sc + ps;
    }
    #pragma unroll
    for (int nt = 0; nt < 4; ++nt)
      #pragma unroll
      for (int r = 0; r < 8; ++r) oacc[nt][r] *= scf[r];

    #pragma unroll
    for (int r = 0; r < 8; ++r) {
      int row = half * 8 + r;
      P[row * 34 + l16]      = f2bf(c2[0][r]);
      P[row * 34 + 16 + l16] = f2bf(c2[1][r]);
    }
    asm volatile("s_wait_dscnt 0" ::: "memory");
    AFrag pf;
    #pragma unroll
    for (int v = 0; v < 8; ++v) {
      int kk2 = (half << 4) + 2 * v;
      pf.u[v] = *(const unsigned*)&P[l16 * 34 + kk2];
    }
    #pragma unroll
    for (int nt = 0; nt < 4; ++nt) {
      AFrag vf;
      int d = nt * 16 + l16;
      #pragma unroll
      for (int v = 0; v < 8; ++v) {
        int key = kb * 32 + (half << 4) + 2 * v;
        int s = (nC - 1) * 256 + key;
        unsigned lo = (s     >= 0) ? vv[((size_t)b * Sc + s)     * 64 + d] : 0u;
        unsigned hi = (s + 1 >= 0) ? vv[((size_t)b * Sc + s + 1) * 64 + d] : 0u;
        vf.u[v] = lo | (hi << 16);
      }
      oacc[nt] = wmma_bf16(pf, vf, oacc[nt]);
    }
  }

  #pragma unroll
  for (int nt = 0; nt < 4; ++nt)
    #pragma unroll
    for (int r = 0; r < 8; ++r) {
      int row = qlo + half * 8 + r;
      int srow = nC * 256 + row;
      int d = nt * 16 + l16;
      out[((size_t)b * Sc + srow) * Dc + h * 64 + d] = f2bf(oacc[nt][r] / lrow[r]);
    }
}

// final combine: out = z*(fwd+bwd) + skip
__global__ void combine_kernel(const float* __restrict__ z,
                               const float* __restrict__ xf,
                               const float* __restrict__ xb,
                               const float* __restrict__ skip,
                               float* __restrict__ out, long n)
{
  long idx = (long)blockIdx.x * 256 + threadIdx.x;
  if (idx >= n) return;
  out[idx] = z[idx >> 8] * (xf[idx] + xb[idx]) + skip[idx];
}

// =======================================================================
// host orchestration
// =======================================================================
static inline void gemm(hipStream_t st, const unsigned short* A, int lda,
                        const unsigned short* W, int ldw, const float* bias,
                        const float* res, int ldres, float* C, int ldc,
                        int M, int N, int K, int act)
{
  dim3 g((N + GBN - 1) / GBN, (M + GBM - 1) / GBM);
  gemm_bf16_wmma<<<g, dim3(256), 0, st>>>(A, lda, W, ldw, bias, res, ldres,
                                          C, ldc, M, N, K, act);
}
static inline long eb(long n) { return (n + 255) / 256; }
static inline void cvt(hipStream_t st, const float* in, unsigned short* o, long n)
{
  cvt_kernel<<<eb(n), 256, 0, st>>>(in, o, n);
}

extern "C" void kernel_launch(void* const* d_in, const int* in_sizes, int n_in,
                              void* d_out, int out_size, void* d_ws, size_t ws_size,
                              hipStream_t stream)
{
  const float* skip   = (const float*)d_in[0];
  const float* ln_g   = (const float*)d_in[1];
  const float* ln_b   = (const float*)d_in[2];
  const float* proj_w = (const float*)d_in[3];
  const float* proj_b = (const float*)d_in[4];
  const float* fconv_w= (const float*)d_in[5];
  const float* fconv_b= (const float*)d_in[6];
  const float* bconv_w= (const float*)d_in[7];
  const float* bconv_b= (const float*)d_in[8];
  const float* rn0_t  = (const float*)d_in[9];
  const float* rn0_c  = (const float*)d_in[10];
  const float* ly_w   = (const float*)d_in[11];
  const float* ly_b   = (const float*)d_in[12];
  const float* lx_w   = (const float*)d_in[13];
  const float* lx_b   = (const float*)d_in[14];
  const float* lo_w   = (const float*)d_in[15];
  const float* lo_b   = (const float*)d_in[16];
  const float* c1d_w  = (const float*)d_in[17];
  const float* c1d_b  = (const float*)d_in[18];
  const float* a_par  = (const float*)d_in[19];
  const float* ig_w   = (const float*)d_in[20];
  const float* ig_b   = (const float*)d_in[21];
  const float* ag_w   = (const float*)d_in[22];
  const float* ag_b   = (const float*)d_in[23];
  const float* m0u_w  = (const float*)d_in[24];
  const float* m0u_b  = (const float*)d_in[25];
  const float* m0d_w  = (const float*)d_in[26];
  const float* m0d_b  = (const float*)d_in[27];
  const float* rn1_t  = (const float*)d_in[28];
  const float* rn1_c  = (const float*)d_in[29];
  const float* q_w    = (const float*)d_in[30];
  const float* k_w    = (const float*)d_in[31];
  const float* v_w    = (const float*)d_in[32];
  const float* o_w    = (const float*)d_in[33];
  const float* o_b    = (const float*)d_in[34];
  const float* m1u_w  = (const float*)d_in[35];
  const float* m1u_b  = (const float*)d_in[36];
  const float* m1d_w  = (const float*)d_in[37];
  const float* m1d_b  = (const float*)d_in[38];
  float* out = (float*)d_out;

  const int   M  = Mrows;
  const long  TD = (long)M * Dc;
  const long  TE = (long)M * Ec;
  const long  TH = (long)M * HDc;
  const long  DD = (long)Dc * Dc;          // 65536
  const long  DE = (long)Dc * Ec;          // 196608
  const long  GG = (long)Hc * HDc * HDc;   // 16384
  const long  DHD= (long)Dc * HDc;         // 16384

  // ---- workspace carve: fp32 region then bf16 region ----
  float* ws = (float*)d_ws;
  size_t off = 0;
  float* zb   = ws + off; off += M;
  float* xf   = ws + off; off += TD;
  float* xb   = ws + off; off += TD;
  float* t0   = ws + off; off += TD;
  float* t1   = ws + off; off += TD;
  float* t2   = ws + off; off += TD;
  float* e0   = ws + off; off += TE;
  float* e1   = ws + off; off += TE;
  float* kbuf = ws + off; off += TH;
  float* vbuf = ws + off; off += TH;
  float* cA   = ws + off; off += (long)Bc * Dc * NCc;
  float* cB   = ws + off; off += (long)Bc * Dc * NCc;
  float* cC   = ws + off; off += (long)Bc * Dc * NCc;

  unsigned short* bws = (unsigned short*)(ws + off);
  size_t bo = 0;
  unsigned short* t0bf = bws + bo; bo += TD;
  unsigned short* t1bf = bws + bo; bo += TD;
  unsigned short* qbf  = bws + bo; bo += TD;
  unsigned short* e0bf = bws + bo; bo += TE;
  unsigned short* kbf  = bws + bo; bo += TH;
  unsigned short* vbf  = bws + bo; bo += TH;
  unsigned short* wproj= bws + bo; bo += DD;
  unsigned short* wfc  = bws + bo; bo += DD;
  unsigned short* wbc  = bws + bo; bo += DD;
  unsigned short* wly  = bws + bo; bo += DD;
  unsigned short* wlx  = bws + bo; bo += DD;
  unsigned short* wlo  = bws + bo; bo += DD;
  unsigned short* wig  = bws + bo; bo += GG;
  unsigned short* wag  = bws + bo; bo += GG;
  unsigned short* wm0u = bws + bo; bo += 2 * DE;
  unsigned short* wm0d = bws + bo; bo += DE;
  unsigned short* wq   = bws + bo; bo += DD;
  unsigned short* wk   = bws + bo; bo += DHD;
  unsigned short* wv   = bws + bo; bo += DHD;
  unsigned short* wo   = bws + bo; bo += DD;
  unsigned short* wm1u = bws + bo; bo += 2 * DE;
  unsigned short* wm1d = bws + bo; bo += DE;
  (void)ws_size; (void)in_sizes; (void)n_in; (void)out_size;

  // ---- convert weights to bf16 once per call ----
  cvt(stream, proj_w, wproj, DD);  cvt(stream, fconv_w, wfc, DD);
  cvt(stream, bconv_w, wbc, DD);   cvt(stream, ly_w, wly, DD);
  cvt(stream, lx_w, wlx, DD);      cvt(stream, lo_w, wlo, DD);
  cvt(stream, ig_w, wig, GG);      cvt(stream, ag_w, wag, GG);
  cvt(stream, m0u_w, wm0u, 2 * DE);cvt(stream, m0d_w, wm0d, DE);
  cvt(stream, q_w, wq, DD);        cvt(stream, k_w, wk, DHD);
  cvt(stream, v_w, wv, DHD);       cvt(stream, o_w, wo, DD);
  cvt(stream, m1u_w, wm1u, 2 * DE);cvt(stream, m1d_w, wm1d, DE);

  // ---- prologue ----
  ln_silu_kernel<<<M, 256, 0, stream>>>(skip, ln_g, ln_b, t0bf, zb);
  gemm(stream, t0bf, Dc, wproj, Dc, proj_b, nullptr, 0, t1, Dc, M, Dc, Dc, 0);
  cvt(stream, t1, t1bf, TD);
  gemm(stream, t1bf, Dc, wfc, Dc, fconv_b, nullptr, 0, xf, Dc, M, Dc, Dc, 0);
  gemm(stream, t1bf, Dc, wbc, Dc, bconv_b, nullptr, 0, xb, Dc, M, Dc, Dc, 0);

  auto run_block = [&](float* X) {
    // ---- ResidualBlock 0: recurrent ----
    rms_kernel<<<M, 256, 0, stream>>>(X, rn0_t, t0bf);
    gemm(stream, t0bf, Dc, wly, Dc, ly_b, nullptr, 0, t1, Dc, M, Dc, Dc, 1);  // y
    gemm(stream, t0bf, Dc, wlx, Dc, lx_b, nullptr, 0, t2, Dc, M, Dc, Dc, 0);  // xr
    conv1d_kernel<<<eb(TD), 256, 0, stream>>>(t2, c1d_w, c1d_b, t0, t0bf, TD);
    for (int h = 0; h < Hc; ++h) {
      gemm(stream, t0bf + h * HDc, Dc, wig + h * HDc * HDc, HDc,
           ig_b + h * HDc, nullptr, 0, t2 + h * HDc, Dc, M, HDc, HDc, 0);
      gemm(stream, t0bf + h * HDc, Dc, wag + h * HDc * HDc, HDc,
           ag_b + h * HDc, nullptr, 0, e0 + h * HDc, Dc, M, HDc, HDc, 0);
    }
    lru_prep_kernel<<<eb(TD), 256, 0, stream>>>(t0, t2, e0, a_par, TD);
    scan1_kernel<<<eb((long)Bc * Dc * NCc), 256, 0, stream>>>(t2, t0, cA, cB);
    scan2_kernel<<<eb((long)Bc * Dc), 256, 0, stream>>>(cA, cB, cC);
    scan3_kernel<<<eb(TD), 256, 0, stream>>>(t2, t0, cC, t1, t0bf, TD);      // h*y
    gemm(stream, t0bf, Dc, wlo, Dc, lo_b, X, Dc, t1, Dc, M, Dc, Dc, 0);      // resid
    rms_kernel<<<M, 256, 0, stream>>>(t1, rn0_c, t0bf);
    gemm(stream, t0bf, Dc, wm0u,      Ec, m0u_b,      nullptr, 0, e0, Ec, M, Ec, Dc, 1);
    gemm(stream, t0bf, Dc, wm0u + DE, Ec, m0u_b + Ec, nullptr, 0, e1, Ec, M, Ec, Dc, 0);
    mul_kernel<<<eb(TE), 256, 0, stream>>>(e0, e1, e0bf, TE);
    gemm(stream, e0bf, Ec, wm0d, Dc, m0d_b, t1, Dc, X, Dc, M, Dc, Ec, 0);    // block0 out

    // ---- ResidualBlock 1: local attention ----
    rms_kernel<<<M, 256, 0, stream>>>(X, rn1_t, t0bf);
    gemm(stream, t0bf, Dc, wq, Dc,  nullptr, nullptr, 0, t2,   Dc,  M, Dc,  Dc, 0);
    gemm(stream, t0bf, Dc, wk, HDc, nullptr, nullptr, 0, kbuf, HDc, M, HDc, Dc, 0);
    gemm(stream, t0bf, Dc, wv, HDc, nullptr, nullptr, 0, vbuf, HDc, M, HDc, Dc, 0);
    rope_bf_kernel<<<eb((long)M * Hc * 16), 256, 0, stream>>>(t2, qbf, Hc, 0.125f, (long)M * Hc * 16);
    rope_bf_kernel<<<eb((long)M * 16), 256, 0, stream>>>(kbuf, kbf, 1, 1.0f, (long)M * 16);
    cvt(stream, vbuf, vbf, TH);
    attn_kernel<<<512, 256, 0, stream>>>(qbf, kbf, vbf, t0bf);
    gemm(stream, t0bf, Dc, wo, Dc, o_b, X, Dc, t1, Dc, M, Dc, Dc, 0);        // resid
    rms_kernel<<<M, 256, 0, stream>>>(t1, rn1_c, t0bf);
    gemm(stream, t0bf, Dc, wm1u,      Ec, m1u_b,      nullptr, 0, e0, Ec, M, Ec, Dc, 1);
    gemm(stream, t0bf, Dc, wm1u + DE, Ec, m1u_b + Ec, nullptr, 0, e1, Ec, M, Ec, Dc, 0);
    mul_kernel<<<eb(TE), 256, 0, stream>>>(e0, e1, e0bf, TE);
    gemm(stream, e0bf, Ec, wm1d, Dc, m1d_b, t1, Dc, X, Dc, M, Dc, Ec, 0);    // block1 out
  };

  run_block(xf);
  run_block(xb);

  combine_kernel<<<eb(TD), 256, 0, stream>>>(zb, xf, xb, skip, out, TD);
}
